// CausalSelfAttention_19559281066074
// MI455X (gfx1250) — matrix operand order repaired and checked by reading, not compiled, not run on hardware
//
#include <hip/hip_runtime.h>
#include <cstdint>

#define T_SEQ 2048
#define NHEAD 16
#define DHEAD 64
#define CDIM  1024

typedef __bf16  bf16x16 __attribute__((ext_vector_type(16)));
typedef float   f32x8   __attribute__((ext_vector_type(8)));
typedef int     i32x4   __attribute__((ext_vector_type(4)));
typedef int     i32x8   __attribute__((ext_vector_type(8)));

union Frag {
    bf16x16 v;
    uint4   u[2];
};

static __device__ __forceinline__ f32x8 wmma_bf16(const Frag& a, const Frag& b, f32x8 c) {
    return __builtin_amdgcn_wmma_f32_16x16x32_bf16(false, a.v, false, b.v, (short)0, c,
                                                   false, false);
}

// Generic pointers to LDS carry the wave-relative LDS byte offset in their low 32 bits.
static __device__ __forceinline__ unsigned lds_lo32(const void* p) {
    return (unsigned)(unsigned long long)p;
}

// Async global->LDS copy, 16 bytes per lane, tracked by ASYNCcnt (CDNA5 path).
static __device__ __forceinline__ void async_copy_b128(const __bf16* g, __bf16* l) {
    asm volatile("global_load_async_to_lds_b128 %0, %1, off"
                 :: "v"(lds_lo32(l)), "v"((unsigned long long)g)
                 : "memory");
}
static __device__ __forceinline__ void wait_async0() {
    asm volatile("s_wait_asynccnt 0" ::: "memory");
}

// Tensor Data Mover: stage a 2D tile (tile_d0 x tile_d1 elements of 2 bytes, row
// stride = `stride` elements) from global to LDS. D# per CDNA5 ISA ch.8.
static __device__ __forceinline__ void tdm_stage_2d_bf16(const __bf16* gbase,
                                                         unsigned lds_addr,
                                                         int tile_d0, int tile_d1,
                                                         int dim0, int dim1, int stride) {
    unsigned long long ga = (unsigned long long)gbase;
    i32x4 g0;
    g0[0] = 1;                                             // count=1 (valid user D#)
    g0[1] = (int)lds_addr;                                 // lds_addr
    g0[2] = (int)(unsigned)(ga & 0xFFFFFFFFull);           // global_addr[31:0]
    g0[3] = (int)((unsigned)((ga >> 32) & 0x01FFFFFFu) | 0x80000000u);  // addr[56:32]|type=2
    i32x8 g1;
    g1[0] = 1 << 16;                                       // data_size=1 (2 bytes)
    g1[1] = (dim0 & 0xFFFF) << 16;                         // tensor_dim0[15:0]
    g1[2] = ((unsigned)dim0 >> 16) | ((dim1 & 0xFFFF) << 16);  // dim0 hi | dim1 lo
    g1[3] = ((unsigned)dim1 >> 16) | ((tile_d0 & 0xFFFF) << 16); // dim1 hi | tile_dim0
    g1[4] = tile_d1 & 0xFFFF;                              // tile_dim1 (tile_dim2=0)
    g1[5] = stride;                                        // tensor_dim0_stride[31:0]
    g1[6] = 0;
    g1[7] = 0;
    asm volatile("tensor_load_to_lds %0, %1" :: "s"(g0), "s"(g1) : "memory");
}

// ---------------------------------------------------------------- conversion
__global__ void cvt_f32_bf16(const float* __restrict__ in, __bf16* __restrict__ out, int n) {
    int i = blockIdx.x * blockDim.x + threadIdx.x;
    int stride = gridDim.x * blockDim.x;
    for (; i < n; i += stride) out[i] = (__bf16)in[i];
}

// out[n*K + k] = (bf16) in[k*N + n]   (weight transpose: [K,N] f32 -> [N,K] bf16)
__global__ void cvt_transpose_bf16(const float* __restrict__ in, __bf16* __restrict__ out,
                                   int K, int N) {
    size_t idx    = (size_t)blockIdx.x * blockDim.x + threadIdx.x;
    size_t total  = (size_t)K * N;
    size_t stride = (size_t)gridDim.x * blockDim.x;
    for (; idx < total; idx += stride) {
        size_t n = idx / K, kk = idx - n * K;
        out[idx] = (__bf16)in[kk * N + n];
    }
}

// ---------------------------------------------------------------- GEMM
// C[M,N] = A[M,K](bf16) @ BT[N,K](bf16, pre-transposed) + bias
// 256 threads (8 waves), tile M=128 x N=128, 8 WMMA per wave per K-step.
// A tile (128x32) staged via the Tensor Data Mover into double-buffered LDS
// (wave 0 issues the descriptor, waits TENSORcnt); B fragments stream straight
// from global (fixed n, contiguous k).  MODE 0: f32 out. MODE 1: qkv scatter.
template <int MODE>
__global__ __launch_bounds__(256) void gemm_bf16(const __bf16* __restrict__ A,
                                                 const __bf16* __restrict__ BT,
                                                 const float* __restrict__ bias,
                                                 int N, int K,
                                                 float* __restrict__ outf,
                                                 __bf16* __restrict__ q_ws,
                                                 __bf16* __restrict__ k_ws,
                                                 __bf16* __restrict__ vT_ws) {
    __shared__ alignas(16) __bf16 a_sh[2][128 * 32];  // double-buffered [m][k]

    const int tid  = threadIdx.x;
    const int lane = tid & 31;
    const int w    = tid >> 5;     // 0..7, wave owns rows w*16..+15
    const int mrow = lane & 15;
    const int half = lane >> 4;
    const int blockM = blockIdx.x * 128;
    const int blockN = blockIdx.y * 128;

    f32x8 acc[8];
    for (int nb = 0; nb < 8; ++nb)
        for (int i = 0; i < 8; ++i) acc[nb][i] = 0.f;

    if (w == 0)  // TDM stage of first A tile
        tdm_stage_2d_bf16(A + (size_t)blockM * K, lds_lo32(&a_sh[0][0]),
                          32, 128, K, 4096, K);

    const int kiters = K >> 5;
    for (int it = 0; it < kiters; ++it) {
        const int k0 = it << 5;
        if (w == 0) __builtin_amdgcn_s_wait_tensorcnt(0);
        __syncthreads();
        if (it + 1 < kiters) {  // TDM-prefetch next A tile into the other buffer
            if (w == 0)
                tdm_stage_2d_bf16(A + (size_t)blockM * K + (k0 + 32),
                                  lds_lo32(&a_sh[(it + 1) & 1][0]),
                                  32, 128, K, 4096, K);
            __builtin_prefetch(&BT[(size_t)(blockN + lane * 4) * K + k0 + 32], 0, 0);
        }
        const __bf16* ab = a_sh[it & 1];
        Frag af;
        af.u[0] = *(const uint4*)&ab[(w * 16 + mrow) * 32 + half * 8];
        af.u[1] = *(const uint4*)&ab[(w * 16 + mrow) * 32 + 16 + half * 8];
#pragma unroll
        for (int nb = 0; nb < 8; ++nb) {
            Frag bfrag;
            const __bf16* bp = &BT[(size_t)(blockN + nb * 16 + mrow) * K + k0 + half * 8];
            bfrag.u[0] = *(const uint4*)bp;
            bfrag.u[1] = *(const uint4*)(bp + 16);
            acc[nb] = wmma_bf16(af, bfrag, acc[nb]);
        }
    }

#pragma unroll
    for (int nb = 0; nb < 8; ++nb) {
#pragma unroll
        for (int r = 0; r < 8; ++r) {
            int m = blockM + w * 16 + r + 8 * half;
            int n = blockN + nb * 16 + mrow;
            float val = acc[nb][r] + bias[n];
            if (MODE == 0) {
                outf[(size_t)m * N + n] = val;
            } else {
                int sect = n >> 10;  // 0:q 1:k 2:v
                int c = n & 1023;
                int h = c >> 6, d = c & 63;
                int b = m >> 11, t = m & 2047;  // T = 2048
                int bh = b * NHEAD + h;
                __bf16 bv = (__bf16)val;
                if (sect == 0)
                    q_ws[((size_t)bh * T_SEQ + t) * DHEAD + d] = bv;
                else if (sect == 1)
                    k_ws[((size_t)bh * T_SEQ + t) * DHEAD + d] = bv;
                else
                    vT_ws[((size_t)bh * DHEAD + d) * T_SEQ + t] = bv;
            }
        }
    }
}

// ---------------------------------------------------------------- attention
static __device__ __forceinline__ f32x8 qk_tile(const Frag* qf, const __bf16* kh,
                                                int keyb, int mrow, int half) {
    f32x8 s;
    for (int i = 0; i < 8; ++i) s[i] = 0.f;
#pragma unroll
    for (int c = 0; c < 2; ++c) {
        Frag kf;
        const __bf16* p = &kh[(size_t)(keyb + mrow) * DHEAD + c * 32 + half * 8];
        kf.u[0] = *(const uint4*)p;
        kf.u[1] = *(const uint4*)(p + 16);
        s = wmma_bf16(qf[c], kf, s);
    }
    return s;
}

__global__ __launch_bounds__(128) void attn_kernel(const __bf16* __restrict__ q,
                                                   const __bf16* __restrict__ k,
                                                   const __bf16* __restrict__ vT,
                                                   __bf16* __restrict__ y) {
    __shared__ alignas(16) __bf16 q_sh[64 * 64];      // block Q tile (async-staged)
    __shared__ alignas(16) __bf16 p_lds[4][16 * 32];  // per-wave P staging

    const int tid  = threadIdx.x;
    const int lane = tid & 31;
    const int w    = tid >> 5;
    const int mrow = lane & 15;
    const int half = lane >> 4;

    const int id = blockIdx.x;
    const int qt = id & 31;  // 32 q-tiles of 64
    const int bh = id >> 5;  // 0..31
    const int b  = bh >> 4, h = bh & 15;

    const __bf16* qh = q  + (size_t)bh * T_SEQ * DHEAD;
    const __bf16* kh = k  + (size_t)bh * T_SEQ * DHEAD;
    const __bf16* vh = vT + (size_t)bh * DHEAD * T_SEQ;

    // async-stage the 64x64 Q tile into LDS (512 x 16B chunks, 4 per thread)
#pragma unroll
    for (int i = 0; i < 4; ++i) {
        int ch  = tid + i * 128;
        int row = ch >> 3, c8 = (ch & 7) * 8;
        async_copy_b128(&qh[(size_t)(qt * 64 + row) * DHEAD + c8], &q_sh[row * 64 + c8]);
    }
    wait_async0();
    __syncthreads();

    const int q0 = qt * 64 + w * 16;

    Frag qf[2];
#pragma unroll
    for (int c = 0; c < 2; ++c) {
        const __bf16* p = &q_sh[(w * 16 + mrow) * 64 + c * 32 + half * 8];
        qf[c].u[0] = *(const uint4*)p;
        qf[c].u[1] = *(const uint4*)(p + 16);
    }

    f32x8 o[4];
    for (int nb = 0; nb < 4; ++nb)
        for (int i = 0; i < 8; ++i) o[nb][i] = 0.f;
    float m_run[8], l_run[8];
    for (int r = 0; r < 8; ++r) { m_run[r] = -1e30f; l_run[r] = 0.f; }

    const int kmax = q0 + 15;
    for (int ktb = 0; ktb <= kmax; ktb += 32) {
        if (ktb + 32 <= kmax) {  // prefetch next K/V tiles
            __builtin_prefetch(&kh[(size_t)(ktb + 32 + lane) * DHEAD], 0, 0);
            __builtin_prefetch(&vh[(size_t)(lane * 2) * T_SEQ + ktb + 32], 0, 0);
        }
        f32x8 s0 = qk_tile(qf, kh, ktb,      mrow, half);
        f32x8 s1 = qk_tile(qf, kh, ktb + 16, mrow, half);

#pragma unroll
        for (int r = 0; r < 8; ++r) {
            int qi = q0 + r + 8 * half;
            float v0 = s0[r] * 0.125f;
            if (ktb + mrow > qi) v0 = -1e30f;
            float v1 = s1[r] * 0.125f;
            if (ktb + 16 + mrow > qi) v1 = -1e30f;
            float mt = fmaxf(v0, v1);
            mt = fmaxf(mt, __shfl_xor(mt, 8, 32));
            mt = fmaxf(mt, __shfl_xor(mt, 4, 32));
            mt = fmaxf(mt, __shfl_xor(mt, 2, 32));
            mt = fmaxf(mt, __shfl_xor(mt, 1, 32));
            float mn = fmaxf(m_run[r], mt);
            float p0 = __expf(v0 - mn);
            float p1 = __expf(v1 - mn);
            p_lds[w][(r + 8 * half) * 32 + mrow]      = (__bf16)p0;
            p_lds[w][(r + 8 * half) * 32 + 16 + mrow] = (__bf16)p1;
            float ls = p0 + p1;
            ls += __shfl_xor(ls, 8, 32);
            ls += __shfl_xor(ls, 4, 32);
            ls += __shfl_xor(ls, 2, 32);
            ls += __shfl_xor(ls, 1, 32);
            float alpha = __expf(m_run[r] - mn);
            l_run[r] = l_run[r] * alpha + ls;
            m_run[r] = mn;
#pragma unroll
            for (int nb = 0; nb < 4; ++nb) o[nb][r] *= alpha;
        }

        // O += P(16x32) @ V(32x64)   (wave-private LDS, in-order DS pipe)
        Frag pf;
        pf.u[0] = *(const uint4*)&p_lds[w][mrow * 32 + half * 8];
        pf.u[1] = *(const uint4*)&p_lds[w][mrow * 32 + 16 + half * 8];
#pragma unroll
        for (int nb = 0; nb < 4; ++nb) {
            Frag vf;
            const __bf16* p = &vh[(size_t)(nb * 16 + mrow) * T_SEQ + ktb + half * 8];
            vf.u[0] = *(const uint4*)p;
            vf.u[1] = *(const uint4*)(p + 16);
            o[nb] = wmma_bf16(pf, vf, o[nb]);
        }
    }

#pragma unroll
    for (int nb = 0; nb < 4; ++nb) {
#pragma unroll
        for (int r = 0; r < 8; ++r) {
            int qrow = q0 + r + 8 * half;
            float val = o[nb][r] / l_run[r];
            y[((size_t)(b * T_SEQ + qrow)) * CDIM + h * DHEAD + nb * 16 + mrow] = (__bf16)val;
        }
    }
}

// ---------------------------------------------------------------- launch
extern "C" void kernel_launch(void* const* d_in, const int* in_sizes, int n_in,
                              void* d_out, int out_size, void* d_ws, size_t ws_size,
                              hipStream_t stream) {
    const float* x      = (const float*)d_in[0];
    const float* W_qkv  = (const float*)d_in[1];
    const float* b_qkv  = (const float*)d_in[2];
    const float* W_proj = (const float*)d_in[3];
    const float* b_proj = (const float*)d_in[4];
    float* out = (float*)d_out;

    char* ws = (char*)d_ws;
    __bf16* x_bf      = (__bf16*)(ws);                          // 8 MB
    __bf16* wqkvT_bf  = (__bf16*)(ws + (size_t)(8u  << 20));    // 6 MB  [3072][1024]
    __bf16* wprojT_bf = (__bf16*)(ws + (size_t)(14u << 20));    // 2 MB  [1024][1024]
    __bf16* q_ws      = (__bf16*)(ws + (size_t)(16u << 20));    // 8 MB
    __bf16* k_ws      = (__bf16*)(ws + (size_t)(24u << 20));    // 8 MB
    __bf16* vT_ws     = (__bf16*)(ws + (size_t)(32u << 20));    // 8 MB
    __bf16* y_bf      = (__bf16*)(ws + (size_t)(40u << 20));    // 8 MB  (total 48 MB)

    cvt_f32_bf16<<<2048, 256, 0, stream>>>(x, x_bf, 4096 * 1024);
    cvt_transpose_bf16<<<4096, 256, 0, stream>>>(W_qkv,  wqkvT_bf,  1024, 3072);
    cvt_transpose_bf16<<<2048, 256, 0, stream>>>(W_proj, wprojT_bf, 1024, 1024);

    dim3 g_qkv(4096 / 128, 3072 / 128);
    gemm_bf16<1><<<g_qkv, 256, 0, stream>>>(x_bf, wqkvT_bf, b_qkv, 3072, 1024,
                                            nullptr, q_ws, k_ws, vT_ws);

    attn_kernel<<<2 * NHEAD * (T_SEQ / 64), 128, 0, stream>>>(q_ws, k_ws, vT_ws, y_bf);

    dim3 g_proj(4096 / 128, 1024 / 128);
    gemm_bf16<0><<<g_proj, 256, 0, stream>>>(y_bf, wprojT_bf, b_proj, 1024, 1024,
                                             out, nullptr, nullptr, nullptr);
}
